// BMN_86457691669066
// MI455X (gfx1250) — matrix operand order repaired and checked by reading, not compiled, not run on hardware
//
#include <hip/hip_runtime.h>

// ---------------- problem constants ----------------
#define T_LEN 100
#define NSAMP 32
#define H1C 256
#define H2C 128
#define H3C 512
#define BATCH 2
#define PIX 10000            // T*T
#define KBIG 3200            // NSAMP*T
#define MBIG 1024            // BATCH*H3

// ---------------- WMMA helpers ----------------
typedef __bf16 bf16x16 __attribute__((ext_vector_type(16)));
typedef float  f32x8   __attribute__((ext_vector_type(8)));
typedef unsigned int u32x4 __attribute__((ext_vector_type(4)));

union Frag {
  u32x4 q[2];
  bf16x16 v;
  unsigned short u[16];
};

__device__ __forceinline__ unsigned short f2bf(float f) {
  unsigned int u = __float_as_uint(f);
  unsigned int r = (u + 0x7FFFu + ((u >> 16) & 1u)) >> 16;   // RNE
  return (unsigned short)r;
}
__device__ __forceinline__ float bf2f(unsigned short s) {
  return __uint_as_float(((unsigned int)s) << 16);
}
__device__ __forceinline__ float sigmoidf_(float x) {
  return 1.0f / (1.0f + __expf(-x));
}

#define BM 64
#define BN 128
#define BK 32
#define LDA 40   // 32 + 8 pad (ushorts); 80B rows keep 16B alignment
#define LDB 40   // B tile stored [n][k] (K-major per column)

// A fragment: 16-bit A 16x32 layout. lane<16: elems 0-7 -> K0..7, 8-15 -> K16..23.
// lane>=16: K8..15 / K24..31.  Two ds_load_b128 per lane.
__device__ __forceinline__ void load_fragA(const unsigned short* sA, int mBase, int lane, Frag& f) {
  int half = lane >> 4, r = lane & 15;
  const unsigned short* p = sA + (mBase + r) * LDA + half * 8;
  f.q[0] = *(const u32x4*)(p);
  f.q[1] = *(const u32x4*)(p + 16);
}
// B fragment: 16-bit B 32x16. lane = column; lane<16 holds K0..7/K16..23, lane>=16 K8..15/K24..31.
__device__ __forceinline__ void load_fragB(const unsigned short* sB, int nBase, int lane, Frag& f) {
  int half = lane >> 4, r = lane & 15;
  const unsigned short* p = sB + (nBase + r) * LDB + half * 8;
  f.q[0] = *(const u32x4*)(p);
  f.q[1] = *(const u32x4*)(p + 16);
}
__device__ __forceinline__ f32x8 wmma_bf16(const Frag& a, const Frag& b, f32x8 c) {
  return __builtin_amdgcn_wmma_f32_16x16x32_bf16(false, a.v, false, b.v, (short)0, c,
                                                 false, false);
}

// =====================================================================
// One-shot mask pack: fp32 sample_mask[t][(n,pix)] -> bf16 BmT[pix][k],
// k = n*100 + t, ld = KBIG.  Reads coalesced over pix; 32B vector stores.
// =====================================================================
__global__ void bmn_mask_pack(const float* __restrict__ mask,
                              unsigned short* __restrict__ BmT) {
  int idx = blockIdx.x * blockDim.x + threadIdx.x;
  const int total = PIX * (KBIG / 16);
  if (idx >= total) return;
  int pix = idx % PIX;
  int kc  = idx / PIX;          // 16-wide k chunk
  int k0  = kc * 16;
  int nidx = k0 / T_LEN;
  int tt   = k0 - nidx * T_LEN;
  union { unsigned short u[16]; u32x4 q[2]; } buf;
#pragma unroll
  for (int i = 0; i < 16; ++i) {
    buf.u[i] = f2bf(mask[(size_t)tt * (NSAMP * PIX) + (size_t)nidx * PIX + pix]);
    if (++tt == T_LEN) { tt = 0; ++nidx; }
  }
  u32x4* dst = (u32x4*)(BmT + (size_t)pix * KBIG + k0);
  dst[0] = buf.q[0];
  dst[1] = buf.q[1];
}

// =====================================================================
// Fused BM GEMM: C[(b,o)][(i,j)] = sum_k A[(b,o)][k] * BmT[(i,j)][k]
//   A:   bf16 [MBIG][KBIG]   (Q projection)
//   BmT: bf16 [PIX][KBIG]    (packed mask, N-major)
//   Block 64x128, 4 waves 2x2, each wave 32x64 (2x4 accums, 8 WMMA/K-step)
//   epilogue: + b3d[o], ReLU, store bf16
// =====================================================================
__global__ void __launch_bounds__(128)
bmn_big_gemm(const unsigned short* __restrict__ A,
             const unsigned short* __restrict__ BmT,
             const float* __restrict__ b3d,
             unsigned short* __restrict__ out) {
  __shared__ __align__(16) unsigned short sA[BM * LDA];
  __shared__ __align__(16) unsigned short sB[BN * LDB];

  const int tid  = threadIdx.x;
  const int lane = tid & 31;
  const int w    = tid >> 5;
  const int wm   = w >> 1, wn = w & 1;
  const int row0 = blockIdx.x * BM;
  const int col0 = blockIdx.y * BN;

  f32x8 zero = {0.f,0.f,0.f,0.f,0.f,0.f,0.f,0.f};
  f32x8 acc[2][4];
#pragma unroll
  for (int i = 0; i < 2; ++i)
#pragma unroll
    for (int j = 0; j < 4; ++j) acc[i][j] = zero;

  // A fill: 64 rows x 32 k / 128 threads (16 contiguous k per thread)
  const int arow = tid >> 1;
  const int ac0  = (tid & 1) * 16;
  // B fill: 128 rows x 32 k / 128 threads (one full K-row per thread)
  // Tail columns (pix >= PIX) load from a clamped address: harmless, since
  // each output column depends only on its own B column and the epilogue
  // guards col < PIX.  No zero-fill, no exec divergence.
  int pixB = col0 + tid;
  if (pixB > PIX - 1) pixB = PIX - 1;
  const unsigned short* gArow = A + (size_t)(row0 + arow) * KBIG + ac0;
  const unsigned short* gBrow = BmT + (size_t)pixB * KBIG;
  u32x4* la = (u32x4*)(sA + arow * LDA + ac0);
  u32x4* lb = (u32x4*)(sB + tid * LDB);

  for (int kb = 0; kb < KBIG; kb += BK) {
    const u32x4* ga = (const u32x4*)(gArow + kb);
    const u32x4* gb = (const u32x4*)(gBrow + kb);
    u32x4 a0v = ga[0], a1v = ga[1];
    u32x4 b0v = gb[0], b1v = gb[1], b2v = gb[2], b3v = gb[3];
    if (kb + BK < KBIG) {
      __builtin_prefetch(gArow + kb + BK, 0, 1);
      __builtin_prefetch(gBrow + kb + BK, 0, 1);
    }
    la[0] = a0v; la[1] = a1v;
    lb[0] = b0v; lb[1] = b1v; lb[2] = b2v; lb[3] = b3v;
    __syncthreads();

    Frag a0, a1, bf[4];
    load_fragA(sA, wm * 32,      lane, a0);
    load_fragA(sA, wm * 32 + 16, lane, a1);
#pragma unroll
    for (int j = 0; j < 4; ++j) load_fragB(sB, wn * 64 + j * 16, lane, bf[j]);
#pragma unroll
    for (int j = 0; j < 4; ++j) {
      acc[0][j] = wmma_bf16(a0, bf[j], acc[0][j]);
      acc[1][j] = wmma_bf16(a1, bf[j], acc[1][j]);
    }
    __syncthreads();
  }

  // ---- epilogue: C/D layout (VGPR v -> M=v+8*half, N=lane&15) ----
  const int half = lane >> 4, r = lane & 15;
#pragma unroll
  for (int i = 0; i < 2; ++i)
#pragma unroll
    for (int j = 0; j < 4; ++j)
#pragma unroll
      for (int v = 0; v < 8; ++v) {
        int row = row0 + wm * 32 + i * 16 + half * 8 + v;
        int col = col0 + wn * 64 + j * 16 + r;
        if (col < PIX) {
          float x = acc[i][j][v] + b3d[row & (H3C - 1)];
          x = fmaxf(x, 0.0f);
          out[(size_t)row * PIX + col] = f2bf(x);
        }
      }
}

// =====================================================================
// Implicit-GEMM 2D conv (1x1 or 3x3, pad=KH/2) via WMMA, batched over z.
//   W:    fp32 [Cout][Cin][KHW]   (bf16-converted in the A loader)
//   in:   bf16 [B][Cin][100][100]
//   out:  bf16 [B][Cout][100][100], bias + ReLU
// Each thread owns one output column (pixel) for the B fill: spatial math
// once per tap, unconditional clamped loads, branch-free mask zeroing.
// =====================================================================
__global__ void __launch_bounds__(128)
bmn_conv2d_wmma(const float* __restrict__ W,
                const unsigned short* __restrict__ in,
                const float* __restrict__ bias,
                unsigned short* __restrict__ out,
                int Cin, int KHW, int Cout) {
  __shared__ __align__(16) unsigned short sA[BM * LDA];
  __shared__ __align__(16) unsigned short sB[BN * LDB];

  const int tid  = threadIdx.x;
  const int lane = tid & 31;
  const int w    = tid >> 5;
  const int wm   = w >> 1, wn = w & 1;
  const int row0 = blockIdx.x * BM;
  const int col0 = blockIdx.y * BN;
  const int b    = blockIdx.z;
  const int Ktot = Cin * KHW;

  f32x8 zero = {0.f,0.f,0.f,0.f,0.f,0.f,0.f,0.f};
  f32x8 acc[2][4];
#pragma unroll
  for (int i = 0; i < 2; ++i)
#pragma unroll
    for (int j = 0; j < 4; ++j) acc[i][j] = zero;

  const int arow = tid >> 1;
  const int ac0  = (tid & 1) * 16;

  // column geometry: one pixel per thread, fixed for the whole kernel
  const int pix  = col0 + tid;
  const int pixc = (pix < PIX) ? pix : (PIX - 1);
  const int oi   = pixc / 100;
  const int oj   = pixc - oi * 100;
  const bool colv = (pix < PIX);

  const unsigned short* inb = in + (size_t)b * Cin * PIX;
  u32x4* la = (u32x4*)(sA + arow * LDA + ac0);
  u32x4* lb = (u32x4*)(sB + tid * LDB);

  int tap = 0, cbase = 0;
  int dh = (KHW == 9) ? -1 : 0;
  int dw = (KHW == 9) ? -1 : 0;

  for (int kb = 0; kb < Ktot; kb += BK) {
    // ---- A tile: weights, strided fp32 gather -> bf16, 2x b128 LDS stores ----
    {
      int o = row0 + arow;               // Cout is a multiple of 64
      const float* wp = W + (size_t)o * Ktot + (size_t)(cbase + ac0) * KHW + tap;
      union { unsigned short u[16]; u32x4 q[2]; } abuf;
#pragma unroll
      for (int i = 0; i < 16; ++i) abuf.u[i] = f2bf(wp[i * KHW]);
      la[0] = abuf.q[0];
      la[1] = abuf.q[1];
    }
    // ---- B tile: 32 channels of one shifted pixel, branch-free padding ----
    {
      int ii = oi + dh, jj = oj + dw;
      bool valid = colv & (ii >= 0) & (ii < 100) & (jj >= 0) & (jj < 100);
      unsigned short msk = valid ? 0xFFFFu : 0u;
      int off = valid ? (ii * 100 + jj) : 0;          // safe address
      const unsigned short* gi = inb + (size_t)cbase * PIX + off;
      union { unsigned short u[32]; u32x4 q[4]; } bbuf;
#pragma unroll
      for (int i = 0; i < 32; ++i)
        bbuf.u[i] = (unsigned short)(gi[(size_t)i * PIX] & msk);
      lb[0] = bbuf.q[0];
      lb[1] = bbuf.q[1];
      lb[2] = bbuf.q[2];
      lb[3] = bbuf.q[3];
    }
    __syncthreads();

    Frag a0, a1, bf[4];
    load_fragA(sA, wm * 32,      lane, a0);
    load_fragA(sA, wm * 32 + 16, lane, a1);
#pragma unroll
    for (int j = 0; j < 4; ++j) load_fragB(sB, wn * 64 + j * 16, lane, bf[j]);
#pragma unroll
    for (int j = 0; j < 4; ++j) {
      acc[0][j] = wmma_bf16(a0, bf[j], acc[0][j]);
      acc[1][j] = wmma_bf16(a1, bf[j], acc[1][j]);
    }
    __syncthreads();

    // advance channel block / tap (BK divides Cin)
    cbase += BK;
    if (cbase >= Cin) {
      cbase = 0;
      ++tap;
      ++dw;
      if (dw > 1) { dw = -1; ++dh; }
    }
  }

  const int half = lane >> 4, r = lane & 15;
#pragma unroll
  for (int i = 0; i < 2; ++i)
#pragma unroll
    for (int j = 0; j < 4; ++j)
#pragma unroll
      for (int v = 0; v < 8; ++v) {
        int row = row0 + wm * 32 + i * 16 + half * 8 + v;
        int col = col0 + wn * 64 + j * 16 + r;
        if (col < PIX) {
          float x = acc[i][j][v] + bias[row];
          x = fmaxf(x, 0.0f);
          out[((size_t)b * Cout + row) * PIX + col] = f2bf(x);
        }
      }
}

// =====================================================================
// Generic grouped conv1d (k=3 or 1), fp32, relu/sigmoid/none.
// in addressed via strides so the (B,T,C) input works without transpose.
// out[b*ob + co*T + t]
// =====================================================================
__global__ void bmn_conv1d(const float* __restrict__ in, int sb, int sc, int st,
                           const float* __restrict__ w, const float* __restrict__ bias,
                           float* __restrict__ out, int ob,
                           int Cin, int Cout, int Gp, int ks, int pad, int act, int total) {
  int idx = blockIdx.x * blockDim.x + threadIdx.x;
  if (idx >= total) return;
  int t  = idx % T_LEN;
  int co = (idx / T_LEN) % Cout;
  int b  = idx / (T_LEN * Cout);
  int cig = Cin / Gp;
  int g   = co / (Cout / Gp);
  float s = bias[co];
  const float* wr = w + (size_t)co * cig * ks;
  const float* ib = in + (size_t)b * sb + (size_t)g * cig * sc;
  for (int ci = 0; ci < cig; ++ci) {
    const float* ic = ib + (size_t)ci * sc;
    const float* wc = wr + ci * ks;
    for (int kk = 0; kk < ks; ++kk) {
      int tt = t + kk - pad;
      if (tt >= 0 && tt < T_LEN) s += wc[kk] * ic[(size_t)tt * st];
    }
  }
  if (act == 0) s = fmaxf(s, 0.0f);
  else if (act == 1) s = sigmoidf_(s);
  out[(size_t)b * ob + (size_t)co * T_LEN + t] = s;
}

// =====================================================================
// Q projection: Q[b,o,n,t] = sum_c w3d[o,c,n] * pf[b,c,t]  -> bf16 A panel
// A layout: [(b*512+o)][n*100+t]
// =====================================================================
__global__ void bmn_qproj(const float* __restrict__ pf,
                          const float* __restrict__ w3d,
                          unsigned short* __restrict__ Abig) {
  int idx = blockIdx.x * blockDim.x + threadIdx.x;
  const int total = BATCH * H3C * NSAMP * T_LEN;
  if (idx >= total) return;
  int t = idx % T_LEN;
  int n = (idx / T_LEN) % NSAMP;
  int o = (idx / (T_LEN * NSAMP)) % H3C;
  int b = idx / (T_LEN * NSAMP * H3C);
  const float* wp = w3d + (size_t)o * (H1C * NSAMP) + n;
  const float* pp = pf + (size_t)b * H1C * T_LEN + t;
  float s = 0.0f;
#pragma unroll 4
  for (int c = 0; c < H1C; ++c)
    s += wp[(size_t)c * NSAMP] * pp[(size_t)c * T_LEN];
  Abig[(size_t)(b * H3C + o) * KBIG + n * T_LEN + t] = f2bf(s);
}

// =====================================================================
// Final 1x1 conv to 2 channels + sigmoid -> fp32 output map
// =====================================================================
__global__ void bmn_head2d(const unsigned short* __restrict__ in,
                           const float* __restrict__ w,
                           const float* __restrict__ bias,
                           float* __restrict__ out) {
  int idx = blockIdx.x * blockDim.x + threadIdx.x;
  const int total = BATCH * PIX;
  if (idx >= total) return;
  int pix = idx % PIX;
  int b   = idx / PIX;
  const unsigned short* gi = in + (size_t)b * H2C * PIX + pix;
  float s0 = bias[0], s1 = bias[1];
  for (int c = 0; c < H2C; ++c) {
    float v = bf2f(gi[(size_t)c * PIX]);
    s0 += w[c] * v;
    s1 += w[H2C + c] * v;
  }
  out[(size_t)b * 2 * PIX + pix]       = sigmoidf_(s0);
  out[(size_t)b * 2 * PIX + PIX + pix] = sigmoidf_(s1);
}

// =====================================================================
extern "C" void kernel_launch(void* const* d_in, const int* in_sizes, int n_in,
                              void* d_out, int out_size, void* d_ws, size_t ws_size,
                              hipStream_t stream) {
  (void)in_sizes; (void)n_in; (void)out_size; (void)ws_size;

  const float* x     = (const float*)d_in[0];
  const float* wb1   = (const float*)d_in[1];
  const float* bb1   = (const float*)d_in[2];
  const float* wb2   = (const float*)d_in[3];
  const float* bb2   = (const float*)d_in[4];
  const float* ws1   = (const float*)d_in[5];
  const float* bs1   = (const float*)d_in[6];
  const float* ws2   = (const float*)d_in[7];
  const float* bs2   = (const float*)d_in[8];
  const float* we1   = (const float*)d_in[9];
  const float* be1   = (const float*)d_in[10];
  const float* we2   = (const float*)d_in[11];
  const float* be2   = (const float*)d_in[12];
  const float* wp    = (const float*)d_in[13];
  const float* bp    = (const float*)d_in[14];
  const float* w3d   = (const float*)d_in[15];
  const float* b3d   = (const float*)d_in[16];
  const float* wq1   = (const float*)d_in[17];
  const float* bq1   = (const float*)d_in[18];
  const float* wq2   = (const float*)d_in[19];
  const float* bq2   = (const float*)d_in[20];
  const float* wq3   = (const float*)d_in[21];
  const float* bq3   = (const float*)d_in[22];
  const float* wq4   = (const float*)d_in[23];
  const float* bq4   = (const float*)d_in[24];
  const float* smask = (const float*)d_in[25];
  float* out = (float*)d_out;

  // ---- workspace carve-up (~107.4 MB) ----
  float* h1 = (float*)d_ws;                     // (2,256,100)
  float* h  = h1 + 51200;
  float* sf = h  + 51200;
  float* ef = sf + 51200;
  float* pf = ef + 51200;
  unsigned short* Abig = (unsigned short*)(pf + 51200);      // [1024][3200]
  unsigned short* BmT  = Abig + (size_t)MBIG * KBIG;         // [10000][3200]
  unsigned short* cm1  = BmT  + (size_t)PIX * KBIG;          // [1024][10000]
  unsigned short* q1o  = cm1  + (size_t)MBIG * PIX;          // [2][128][10000]
  unsigned short* q2o  = q1o  + (size_t)BATCH * H2C * PIX;
  unsigned short* q3o  = q2o  + (size_t)BATCH * H2C * PIX;

  const int tot1 = BATCH * H1C * T_LEN;         // 51200
  const int totH = BATCH * T_LEN;               // 200

  // mask pack first: independent of the conv stack
  {
    const int tot = PIX * (KBIG / 16);          // 2,000,000
    bmn_mask_pack<<<(tot + 255) / 256, 256, 0, stream>>>(smask, BmT);
  }

  // 1-D conv stack (small; plain VALU)
  bmn_conv1d<<<(tot1 + 255) / 256, 256, 0, stream>>>(
      x, T_LEN * 400, 1, 400, wb1, bb1, h1, H1C * T_LEN, 400, H1C, 4, 3, 1, 0, tot1);
  bmn_conv1d<<<(tot1 + 255) / 256, 256, 0, stream>>>(
      h1, H1C * T_LEN, T_LEN, 1, wb2, bb2, h, H1C * T_LEN, H1C, H1C, 4, 3, 1, 0, tot1);
  bmn_conv1d<<<(tot1 + 255) / 256, 256, 0, stream>>>(
      h, H1C * T_LEN, T_LEN, 1, ws1, bs1, sf, H1C * T_LEN, H1C, H1C, 4, 3, 1, 0, tot1);
  bmn_conv1d<<<(totH + 255) / 256, 256, 0, stream>>>(
      sf, H1C * T_LEN, T_LEN, 1, ws2, bs2, out + 4 * PIX, T_LEN, H1C, 1, 1, 1, 0, 1, totH);
  bmn_conv1d<<<(tot1 + 255) / 256, 256, 0, stream>>>(
      h, H1C * T_LEN, T_LEN, 1, we1, be1, ef, H1C * T_LEN, H1C, H1C, 4, 3, 1, 0, tot1);
  bmn_conv1d<<<(totH + 255) / 256, 256, 0, stream>>>(
      ef, H1C * T_LEN, T_LEN, 1, we2, be2, out + 4 * PIX + totH, T_LEN, H1C, 1, 1, 1, 0, 1, totH);
  bmn_conv1d<<<(tot1 + 255) / 256, 256, 0, stream>>>(
      h, H1C * T_LEN, T_LEN, 1, wp, bp, pf, H1C * T_LEN, H1C, H1C, 1, 3, 1, 0, tot1);

  // Q projection -> bf16 A panel for the fused BM GEMM
  {
    const int tot = BATCH * H3C * NSAMP * T_LEN;   // 3,276,800
    bmn_qproj<<<(tot + 255) / 256, 256, 0, stream>>>(pf, w3d, Abig);
  }

  // Fused BM-layer + Conv3d GEMM: 1024 x 10000 x 3200 (bf16 WMMA)
  {
    dim3 grid(MBIG / BM, (PIX + BN - 1) / BN, 1);  // 16 x 79
    bmn_big_gemm<<<grid, 128, 0, stream>>>(Abig, BmT, b3d, cm1);
  }

  // 2-D conv tower (implicit-GEMM WMMA, bf16 activations end-to-end)
  {
    dim3 grid(H2C / BM, (PIX + BN - 1) / BN, BATCH);  // 2 x 79 x 2
    bmn_conv2d_wmma<<<grid, 128, 0, stream>>>(wq1, cm1, bq1, q1o, H3C, 1, H2C);
    bmn_conv2d_wmma<<<grid, 128, 0, stream>>>(wq2, q1o, bq2, q2o, H2C, 9, H2C);
    bmn_conv2d_wmma<<<grid, 128, 0, stream>>>(wq3, q2o, bq3, q3o, H2C, 9, H2C);
  }

  // Final 2-channel 1x1 + sigmoid -> fp32 output maps
  {
    const int tot = BATCH * PIX;
    bmn_head2d<<<(tot + 255) / 256, 256, 0, stream>>>(q3o, wq4, bq4, out);
  }
}
